// Head_77120432767536
// MI455X (gfx1250) — matrix operand order repaired and checked by reading, not compiled, not run on hardware
//
#include <hip/hip_runtime.h>
#include <hip/hip_bf16.h>

#define DIM 1024
#define HEAD 128
#define BATCH 4
#define SEQ 4096

typedef __bf16 v16bf __attribute__((ext_vector_type(16)));
typedef float  v8f   __attribute__((ext_vector_type(8)));

__device__ __forceinline__ v8f wmma_bf16(v16bf a, v16bf b, v8f c) {
    // D(f32 16x16) = A(bf16 16x32) * B(bf16 32x16) + C
    return __builtin_amdgcn_wmma_f32_16x16x32_bf16(
        /*neg_a=*/false, a, /*neg_b=*/false, b,
        /*c_mod=*/(short)0, c, /*reuse_a=*/false, /*reuse_b=*/false);
}

// A-fragment load: 16x32 bf16 tile, row = lane&15, K-half split by lane half.
__device__ __forceinline__ void load_afrag(const __bf16* __restrict__ p,
                                           int kb, v16bf& a) {
    #pragma unroll
    for (int j = 0; j < 8; ++j) {
        a[j]     = p[kb + j];
        a[j + 8] = p[kb + 16 + j];
    }
}
// B-fragment load: 2 runs of 8 bf16 per lane (p[0..7], p[16..23]).
__device__ __forceinline__ void load_bfrag(const __bf16* __restrict__ p, v16bf& b) {
    #pragma unroll
    for (int j = 0; j < 8; ++j) { b[j] = p[j]; b[j + 8] = p[16 + j]; }
}

// ---------------------------------------------------------------------------
// Kernel 0: one-shot f32 -> bf16 conversion (8 elements / thread, b128 traffic)
// ---------------------------------------------------------------------------
__global__ void __launch_bounds__(256) cvt_bf16(const float* __restrict__ in,
                                                __bf16* __restrict__ out,
                                                int n8) {
    const int t = blockIdx.x * blockDim.x + threadIdx.x;
    if (t < n8) {
        const float* p = in + (long)t * 8;
        __bf16* o = out + (long)t * 8;
        #pragma unroll
        for (int j = 0; j < 8; ++j) o[j] = (__bf16)p[j];
    }
}

// ---------------------------------------------------------------------------
// Projection helper: one 16-token tile x one weight matrix (all bf16).
// 4-deep rolling B-fragment pipeline: at step hc, issue the load of fragment
// (hc+3) into b[(hc+3)&3] and WMMA with b[hc&3] -> 3 fragments always in
// flight, every buffer index a compile-time constant (no spills).
// The lookahead rolls over the kc-pair bodies (8 == 0 mod 4); tail lookahead
// stays inside W, and the one-ahead A prefetch reads into the padded x buffer.
// TRANSPOSE=false: out[r*HEAD + h]   [Q, K]
// TRANSPOSE=true : out[h*SEQ + r]    [V^T per batch]
// ---------------------------------------------------------------------------
template <bool TRANSPOSE>
__device__ __forceinline__ void proj_one(const __bf16* __restrict__ xrow,
                                         const __bf16* __restrict__ W,
                                         __bf16* __restrict__ out,
                                         int row_in, int half, float scale) {
    const __bf16* wbase = W + (long)row_in * DIM + half * 8;
    auto bptr = [&](int kcc, int hc) {
        return wbase + (long)hc * (16 * DIM) + kcc * 32;
    };
    v8f acc[8] = {};
    v16bf a0, a1, b[4];
    load_afrag(xrow, half * 8, a0);
    load_bfrag(bptr(0, 0), b[0]);
    load_bfrag(bptr(0, 1), b[1]);
    load_bfrag(bptr(0, 2), b[2]);

    for (int kc = 0; kc < DIM / 32; kc += 2) {
        load_afrag(xrow, (kc + 1) * 32 + half * 8, a1);      // prefetch A(kc+1)
        #pragma unroll
        for (int hc = 0; hc < 8; ++hc) {
            const __bf16* nb = (hc < 5) ? bptr(kc, hc + 3) : bptr(kc + 1, hc - 5);
            load_bfrag(nb, b[(hc + 3) & 3]);                 // 3 frags in flight
            acc[hc] = wmma_bf16(a0, b[hc & 3], acc[hc]);
        }
        load_afrag(xrow, (kc + 2) * 32 + half * 8, a0);      // prefetch A(kc+2), padded
        #pragma unroll
        for (int hc = 0; hc < 8; ++hc) {
            const __bf16* nb = (hc < 5) ? bptr(kc + 1, hc + 3) : bptr(kc + 2, hc - 5);
            load_bfrag(nb, b[(hc + 3) & 3]);
            acc[hc] = wmma_bf16(a1, b[hc & 3], acc[hc]);
        }
    }
    #pragma unroll
    for (int hc = 0; hc < 8; ++hc) {
        const int h = hc * 16 + row_in;
        #pragma unroll
        for (int i = 0; i < 8; ++i) {
            const int r = i + half * 8;
            if constexpr (TRANSPOSE)
                out[(long)h * SEQ + r] = (__bf16)(acc[hc][i] * scale);
            else
                out[(long)r * HEAD + h] = (__bf16)(acc[hc][i] * scale);
        }
    }
}

// One wave per 16-token tile; 8 waves / block.
__global__ void __launch_bounds__(256) qkv_proj(
        const __bf16* __restrict__ x,   // [B*SEQ, DIM] bf16 (padded by 64 elems)
        const __bf16* __restrict__ Wq,  // [HEAD, DIM] bf16
        const __bf16* __restrict__ Wk,
        const __bf16* __restrict__ Wv,
        __bf16* __restrict__ qo,        // [B*SEQ, HEAD], pre-scaled by 1/32 (exact)
        __bf16* __restrict__ ko,        // [B*SEQ, HEAD]
        __bf16* __restrict__ vt) {      // [B, HEAD, SEQ]  (transposed)
    const int lane   = threadIdx.x & 31;
    const int wave   = threadIdx.x >> 5;
    const int row_in = lane & 15;
    const int half   = (lane >> 4) & 1;
    const long tok   = ((long)blockIdx.x * 8 + wave) * 16;   // over B*SEQ tokens
    const long bb    = tok / SEQ;                            // tile never straddles batch
    const long nbase = tok % SEQ;
    const __bf16* xrow = x + (tok + row_in) * DIM;

    // 1024^-0.5 == 1/32: power of two -> folding into bf16 Q store is exact
    proj_one<false>(xrow, Wq, qo + tok * HEAD,              row_in, half, 0.03125f);
    proj_one<false>(xrow, Wk, ko + tok * HEAD,              row_in, half, 1.0f);
    proj_one<true >(xrow, Wv, vt + bb * HEAD * SEQ + nbase, row_in, half, 1.0f);
}

// ---------------------------------------------------------------------------
// Kernel 2: streaming (flash) attention.  One wave = 16 query rows.
// Per 32-key tile: bulk-load 8 K-frags -> 8 S-WMMAs; then issue all 8 V-frag
// loads BEFORE the softmax so exp/shuffle VALU and the LDS P-transpose hide
// the V load latency; finally 8 PV-WMMAs.
// ---------------------------------------------------------------------------
__global__ void __launch_bounds__(128) flash_attn(
        const __bf16* __restrict__ q,   // pre-scaled
        const __bf16* __restrict__ k,
        const __bf16* __restrict__ vt,  // [B, HEAD, SEQ]
        float* __restrict__ out) {
    __shared__ __bf16 pbuf[4][16][32];                       // 4 KB, per-wave slice

    const int lane   = threadIdx.x & 31;
    const int wave   = threadIdx.x >> 5;
    const int bb     = blockIdx.y;
    const int qbase  = blockIdx.x * 64 + wave * 16;          // query row within batch
    const int row_in = lane & 15;
    const int half   = (lane >> 4) & 1;
    const long base  = (long)bb * SEQ * HEAD;                // for q/k/out
    const long vbase = (long)bb * HEAD * SEQ;                // for vt

    // Q fragments: 4 chunks of K=32 over head dim 128 (held in VGPRs all loop)
    v16bf qa[4];
    {
        const __bf16* qrow = q + base + (long)(qbase + row_in) * HEAD;
        #pragma unroll
        for (int kc = 0; kc < 4; ++kc)
            load_afrag(qrow, kc * 32 + half * 8, qa[kc]);
    }

    v8f o[8] = {};                                           // 16x128 f32 output accum
    float m_i[8], l_i[8];
    #pragma unroll
    for (int i = 0; i < 8; ++i) { m_i[i] = -3.0e38f; l_i[i] = 0.0f; }

    for (int kt = 0; kt < SEQ; kt += 32) {
        // prefetch next key/value tiles into cache (global_prefetch_b8)
        if (kt + 32 < SEQ) {
            __builtin_prefetch(k + base + (long)(kt + 32 + row_in) * HEAD);
            __builtin_prefetch(vt + vbase + (long)(row_in * 8) * SEQ + kt + 32);
        }
        // ---- S = Q * K^T for 32 keys: bulk-load 8 K-frags, then WMMA chain ----
        const __bf16* krow = k + base + (long)(kt + row_in) * HEAD + half * 8;
        v16bf kf[8];
        #pragma unroll
        for (int kc = 0; kc < 4; ++kc) {
            load_bfrag(krow + kc * 32, kf[kc * 2]);
            load_bfrag(krow + kc * 32 + 16 * HEAD, kf[kc * 2 + 1]);
        }
        v8f s0 = {}, s1 = {};
        #pragma unroll
        for (int kc = 0; kc < 4; ++kc) {
            s0 = wmma_bf16(qa[kc], kf[kc * 2], s0);
            s1 = wmma_bf16(qa[kc], kf[kc * 2 + 1], s1);
        }
        // ---- issue all V-frag loads now; softmax + LDS hide their latency ----
        const __bf16* vrow = vt + vbase + (long)row_in * SEQ + kt + half * 8;
        v16bf vf[8];
        #pragma unroll
        for (int hc = 0; hc < 8; ++hc)
            load_bfrag(vrow + (long)hc * 16 * SEQ, vf[hc]);
        // ---- online softmax (Q pre-scaled; C layout row stats) ----
        float p0[8], p1[8], corr[8], rm[8], rs[8];
        #pragma unroll
        for (int i = 0; i < 8; ++i) rm[i] = fmaxf(s0[i], s1[i]);
        #pragma unroll
        for (int mask = 1; mask < 16; mask <<= 1) {
            #pragma unroll
            for (int i = 0; i < 8; ++i)
                rm[i] = fmaxf(rm[i], __shfl_xor(rm[i], mask, 32));
        }
        #pragma unroll
        for (int i = 0; i < 8; ++i) {
            const float mn = fmaxf(m_i[i], rm[i]);
            corr[i] = __expf(m_i[i] - mn);
            m_i[i]  = mn;
            p0[i]   = __expf(s0[i] - mn);
            p1[i]   = __expf(s1[i] - mn);
            rs[i]   = p0[i] + p1[i];
        }
        #pragma unroll
        for (int mask = 1; mask < 16; mask <<= 1) {
            #pragma unroll
            for (int i = 0; i < 8; ++i)
                rs[i] += __shfl_xor(rs[i], mask, 32);
        }
        #pragma unroll
        for (int i = 0; i < 8; ++i) l_i[i] = l_i[i] * corr[i] + rs[i];
        #pragma unroll
        for (int hc = 0; hc < 8; ++hc) {
            #pragma unroll
            for (int i = 0; i < 8; ++i) o[hc][i] *= corr[i];
        }
        // ---- transpose P (C layout -> A layout) through per-wave LDS ----
        #pragma unroll
        for (int i = 0; i < 8; ++i) {
            const int r = i + half * 8;
            pbuf[wave][r][row_in]      = (__bf16)p0[i];
            pbuf[wave][r][16 + row_in] = (__bf16)p1[i];
        }
        asm volatile("s_wait_dscnt 0x0" ::: "memory");       // wave-private: no barrier
        v16bf pa;
        {
            const int kb = half * 8;
            #pragma unroll
            for (int j = 0; j < 8; ++j) {
                pa[j]     = pbuf[wave][row_in][kb + j];
                pa[j + 8] = pbuf[wave][row_in][kb + 16 + j];
            }
        }
        // ---- O += P * V ----
        #pragma unroll
        for (int hc = 0; hc < 8; ++hc)
            o[hc] = wmma_bf16(pa, vf[hc], o[hc]);
    }

    // ---- normalize and store f32 context ----
    float* op = out + base + (long)qbase * HEAD;
    #pragma unroll
    for (int i = 0; i < 8; ++i) {
        const float inv = 1.0f / l_i[i];
        const int r = i + half * 8;
        #pragma unroll
        for (int hc = 0; hc < 8; ++hc)
            op[(long)r * HEAD + hc * 16 + row_in] = o[hc][i] * inv;
    }
}

extern "C" void kernel_launch(void* const* d_in, const int* in_sizes, int n_in,
                              void* d_out, int out_size, void* d_ws, size_t ws_size,
                              hipStream_t stream) {
    const float* x  = (const float*)d_in[0];
    const float* Wq = (const float*)d_in[1];
    const float* Wk = (const float*)d_in[2];
    const float* Wv = (const float*)d_in[3];
    float* out = (float*)d_out;

    const size_t qkv_elems = (size_t)BATCH * SEQ * HEAD;     // 2M elems each
    const size_t x_elems   = (size_t)BATCH * SEQ * DIM;      // 16M elems
    const size_t w_elems   = (size_t)HEAD * DIM;             // 128K elems
    const size_t xpad      = 64;                             // A-prefetch tail pad

    __bf16* qb  = (__bf16*)d_ws;
    __bf16* kb  = qb + qkv_elems;
    __bf16* vb  = kb + qkv_elems;        // 12 MB
    __bf16* xb  = vb + qkv_elems;        // +32 MB (+pad)
    __bf16* wqb = xb + x_elems + xpad;   // +256 KB x3
    __bf16* wkb = wqb + w_elems;
    __bf16* wvb = wkb + w_elems;

    // ---- one-shot bf16 conversion of x and the three weight matrices ----
    {
        const int n8x = (int)(x_elems / 8);
        cvt_bf16<<<dim3((n8x + 255) / 256), dim3(256), 0, stream>>>(x, xb, n8x);
        const int n8w = (int)(w_elems / 8);
        cvt_bf16<<<dim3((n8w + 255) / 256), dim3(256), 0, stream>>>(Wq, wqb, n8w);
        cvt_bf16<<<dim3((n8w + 255) / 256), dim3(256), 0, stream>>>(Wk, wkb, n8w);
        cvt_bf16<<<dim3((n8w + 255) / 256), dim3(256), 0, stream>>>(Wv, wvb, n8w);
    }

    // 1024 token tiles of 16 rows, 8 waves (tiles) per 256-thread block
    qkv_proj<<<dim3((BATCH * SEQ / 16) / 8), dim3(256), 0, stream>>>(
        xb, wqb, wkb, wvb, qb, kb, vb);

    // 4 waves/block * 16 query rows = 64 rows per block; grid = (SEQ/64, B)
    flash_attn<<<dim3(SEQ / 64, BATCH), dim3(128), 0, stream>>>(
        qb, kb, vb, out);
}
